// SelfAttention_81492709474650
// MI455X (gfx1250) — matrix operand order repaired and checked by reading, not compiled, not run on hardware
//
#include <hip/hip_runtime.h>
#include <hip/hip_bf16.h>

// ---------------------------------------------------------------------------
// Self-attention forward for MI455X (gfx1250), bf16 WMMA pipeline:
//   1) cast X, Wq, Wk, Wv  f32 -> bf16            (workspace)
//   2) Q/K/V = X @ W^T + b          (WMMA GEMM, B given as [N,K])
//   3) S = (Q @ K^T) * 1/sqrt(H)    (WMMA GEMM, B given as [N,K]) -> bf16
//   4) P = softmax(mask(S))         (row kernel, in place, bf16)
//   5) out = P @ V                  (WMMA GEMM, B given as [K,N]) -> f32
// GEMM core: 128x128 macro tile, K-step 32, double-buffered LDS with ONE
// barrier per k-step; global->LDS staging uses the CDNA5 async-to-LDS path
// (ASYNCcnt) when the builtin is available, register staging otherwise.
// ---------------------------------------------------------------------------

typedef __attribute__((ext_vector_type(16))) __bf16 v16bf;
typedef __attribute__((ext_vector_type(8)))  __bf16 v8bf;
typedef __attribute__((ext_vector_type(8)))  float  v8f;
typedef __attribute__((ext_vector_type(4)))  int    v4i;

#define BDIM 256

static constexpr int kB = 2;
static constexpr int kS = 4096;
static constexpr int kE = 1024;
static constexpr int kH = 1024;

// ---- async global->LDS support (guarded; falls back to register staging) --
#define TRY_ASYNC_LDS 1
#if TRY_ASYNC_LDS && defined(__has_builtin)
#if __has_builtin(__builtin_amdgcn_global_load_async_to_lds_b128)
#define USE_ASYNC_LDS 1
#endif
#endif
#ifndef USE_ASYNC_LDS
#define USE_ASYNC_LDS 0
#endif

#if USE_ASYNC_LDS
__device__ __forceinline__ void g2l_async16(const __bf16* g, __bf16* l) {
  __builtin_amdgcn_global_load_async_to_lds_b128(
      (__attribute__((address_space(1))) v4i*)(void*)g,
      (__attribute__((address_space(3))) v4i*)(void*)l, 0, 0);
}
#endif

__device__ __forceinline__ void wait_async_lds() {
#if USE_ASYNC_LDS
#if defined(__has_builtin)
#if __has_builtin(__builtin_amdgcn_s_wait_asynccnt)
  __builtin_amdgcn_s_wait_asynccnt(0);
#else
  asm volatile("s_wait_asynccnt 0x0" ::: "memory");
#endif
#else
  asm volatile("s_wait_asynccnt 0x0" ::: "memory");
#endif
#endif
}

// ---------------------------------------------------------------------------
__global__ __launch_bounds__(BDIM)
void cast_f32_to_bf16(const float* __restrict__ in, __bf16* __restrict__ out,
                      long long n) {
  long long i = (long long)blockIdx.x * BDIM + threadIdx.x;
  const long long stride = (long long)gridDim.x * BDIM;
  for (; i < n; i += stride) out[i] = (__bf16)in[i];
}

// ---------------------------------------------------------------------------
// Tiled bf16 WMMA GEMM:  O[M,N] = scale * (A[M,K] x B') + bias
//   BT == true : B source is [N,K] row-major (compute A x Bsrc^T)
//   BT == false: B source is [K,N] row-major (scatter-transposed into LDS)
// 8 waves (4 in M x 2 in N), wave tile 32x64, 8 f32 accum tiles per wave.
// ---------------------------------------------------------------------------
template <bool BT, typename OutT>
__global__ __launch_bounds__(BDIM)
void wmma_gemm_bf16(const __bf16* __restrict__ A, const __bf16* __restrict__ Bm,
                    OutT* __restrict__ O, const float* __restrict__ bias,
                    float scale, int M, int N, int K,
                    long long sA, long long sB, long long sO) {
  __shared__ __align__(16) __bf16 Asm[2][128][40];  // [buf][m][k], padded
  __shared__ __align__(16) __bf16 Bsm[2][128][40];  // [buf][n][k], padded

  const __bf16* __restrict__ Ab = A + (long long)blockIdx.z * sA;
  const __bf16* __restrict__ Bb = Bm + (long long)blockIdx.z * sB;
  OutT* __restrict__ Ob = O + (long long)blockIdx.z * sO;

  const int tid   = threadIdx.x;
  const int lane  = tid & 31;
  const int wave  = tid >> 5;
  const int waveM = wave & 3;   // 0..3 -> 4 * 32 rows
  const int waveN = wave >> 2;  // 0..1 -> 2 * 64 cols
  const int lh    = lane >> 4;  // half-wave select
  const int lr    = lane & 15;
  const int mBlock = blockIdx.y * 128;
  const int nBlock = blockIdx.x * 128;

  // per-thread staging coordinates (two 16B chunks per tile per thread)
  const int srow = tid >> 2;         // 0..63  (rows srow, srow+64)
  const int scol = (tid & 3) * 8;    // 0,8,16,24
  const int bkk  = tid >> 4;         // 0..15  (non-BT k rows bkk, bkk+16)
  const int bnn  = (tid & 15) * 8;   // 0..120 (non-BT n chunk)

  v8f acc[2][4];
#pragma unroll
  for (int m = 0; m < 2; ++m)
#pragma unroll
    for (int n = 0; n < 4; ++n) acc[m][n] = (v8f)0.0f;

  const int nk = K >> 5;

  // ---- prologue: stage tile 0 into buffer 0 ----
  {
#if USE_ASYNC_LDS
    g2l_async16(Ab + (long long)(mBlock + srow) * K + scol, &Asm[0][srow][scol]);
    g2l_async16(Ab + (long long)(mBlock + srow + 64) * K + scol,
                &Asm[0][srow + 64][scol]);
    if (BT) {
      g2l_async16(Bb + (long long)(nBlock + srow) * K + scol,
                  &Bsm[0][srow][scol]);
      g2l_async16(Bb + (long long)(nBlock + srow + 64) * K + scol,
                  &Bsm[0][srow + 64][scol]);
    } else {
      v8bf r0 = *(const v8bf*)(Bb + (long long)bkk * N + nBlock + bnn);
      v8bf r1 = *(const v8bf*)(Bb + (long long)(bkk + 16) * N + nBlock + bnn);
#pragma unroll
      for (int e = 0; e < 8; ++e) {
        Bsm[0][bnn + e][bkk] = r0[e];
        Bsm[0][bnn + e][bkk + 16] = r1[e];
      }
    }
    wait_async_lds();
#else
    *(v8bf*)&Asm[0][srow][scol] =
        *(const v8bf*)(Ab + (long long)(mBlock + srow) * K + scol);
    *(v8bf*)&Asm[0][srow + 64][scol] =
        *(const v8bf*)(Ab + (long long)(mBlock + srow + 64) * K + scol);
    if (BT) {
      *(v8bf*)&Bsm[0][srow][scol] =
          *(const v8bf*)(Bb + (long long)(nBlock + srow) * K + scol);
      *(v8bf*)&Bsm[0][srow + 64][scol] =
          *(const v8bf*)(Bb + (long long)(nBlock + srow + 64) * K + scol);
    } else {
      v8bf r0 = *(const v8bf*)(Bb + (long long)bkk * N + nBlock + bnn);
      v8bf r1 = *(const v8bf*)(Bb + (long long)(bkk + 16) * N + nBlock + bnn);
#pragma unroll
      for (int e = 0; e < 8; ++e) {
        Bsm[0][bnn + e][bkk] = r0[e];
        Bsm[0][bnn + e][bkk + 16] = r1[e];
      }
    }
#endif
    __syncthreads();
  }

  // ---- main loop: one barrier per k-step, read buf / write buf^1 ----
  for (int kt = 0; kt < nk; ++kt) {
    const int buf = kt & 1;
    const bool more = (kt + 1) < nk;
    const int kn = (kt + 1) << 5;

    v8bf rb0 = {}, rb1 = {};          // non-BT pending register tile
#if !USE_ASYNC_LDS
    v8bf ra0 = {}, ra1 = {}, rc0 = {}, rc1 = {};
#endif
    if (more) {
      // L2 prefetch two tiles ahead
      if (kt + 2 < nk) {
        __builtin_prefetch(Ab + (long long)(mBlock + srow) * K + kn + 32 + scol,
                           0, 1);
      }
#if USE_ASYNC_LDS
      g2l_async16(Ab + (long long)(mBlock + srow) * K + kn + scol,
                  &Asm[buf ^ 1][srow][scol]);
      g2l_async16(Ab + (long long)(mBlock + srow + 64) * K + kn + scol,
                  &Asm[buf ^ 1][srow + 64][scol]);
      if (BT) {
        g2l_async16(Bb + (long long)(nBlock + srow) * K + kn + scol,
                    &Bsm[buf ^ 1][srow][scol]);
        g2l_async16(Bb + (long long)(nBlock + srow + 64) * K + kn + scol,
                    &Bsm[buf ^ 1][srow + 64][scol]);
      } else {
        rb0 = *(const v8bf*)(Bb + (long long)(kn + bkk) * N + nBlock + bnn);
        rb1 = *(const v8bf*)(Bb + (long long)(kn + bkk + 16) * N + nBlock + bnn);
      }
#else
      ra0 = *(const v8bf*)(Ab + (long long)(mBlock + srow) * K + kn + scol);
      ra1 = *(const v8bf*)(Ab + (long long)(mBlock + srow + 64) * K + kn + scol);
      if (BT) {
        rc0 = *(const v8bf*)(Bb + (long long)(nBlock + srow) * K + kn + scol);
        rc1 = *(const v8bf*)(Bb + (long long)(nBlock + srow + 64) * K + kn + scol);
      } else {
        rb0 = *(const v8bf*)(Bb + (long long)(kn + bkk) * N + nBlock + bnn);
        rb1 = *(const v8bf*)(Bb + (long long)(kn + bkk + 16) * N + nBlock + bnn);
      }
#endif
    }

    // ---- fragments from current buffer ----
    // A (16x32 bf16): lanes 0-15 K{0..7,16..23}; lanes 16-31 K{8..15,24..31}
    v16bf af[2];
#pragma unroll
    for (int m = 0; m < 2; ++m) {
      const __bf16* rp = &Asm[buf][waveM * 32 + m * 16 + lr][0];
      v8bf c0 = *(const v8bf*)(rp + lh * 8);
      v8bf c1 = *(const v8bf*)(rp + lh * 8 + 16);
      af[m] = __builtin_shufflevector(c0, c1, 0, 1, 2, 3, 4, 5, 6, 7,
                                      8, 9, 10, 11, 12, 13, 14, 15);
    }
    // B (32x16 bf16): lane n = lr; lanes 0-15 K{0..7,16..23}; lanes 16-31 K{8..15,24..31}
    v16bf bfv[4];
#pragma unroll
    for (int n = 0; n < 4; ++n) {
      const __bf16* rp = &Bsm[buf][waveN * 64 + n * 16 + lr][0];
      v8bf c0 = *(const v8bf*)(rp + lh * 8);
      v8bf c1 = *(const v8bf*)(rp + lh * 8 + 16);
      bfv[n] = __builtin_shufflevector(c0, c1, 0, 1, 2, 3, 4, 5, 6, 7,
                                       8, 9, 10, 11, 12, 13, 14, 15);
    }

#pragma unroll
    for (int m = 0; m < 2; ++m)
#pragma unroll
      for (int n = 0; n < 4; ++n)
        acc[m][n] = __builtin_amdgcn_wmma_f32_16x16x32_bf16(
            false, af[m], false, bfv[n], (short)0, acc[m][n], false, false);

    // ---- complete next-tile staging into buf^1 ----
    if (more) {
#if USE_ASYNC_LDS
      if (!BT) {
#pragma unroll
        for (int e = 0; e < 8; ++e) {
          Bsm[buf ^ 1][bnn + e][bkk] = rb0[e];
          Bsm[buf ^ 1][bnn + e][bkk + 16] = rb1[e];
        }
      }
      wait_async_lds();
#else
      *(v8bf*)&Asm[buf ^ 1][srow][scol] = ra0;
      *(v8bf*)&Asm[buf ^ 1][srow + 64][scol] = ra1;
      if (BT) {
        *(v8bf*)&Bsm[buf ^ 1][srow][scol] = rc0;
        *(v8bf*)&Bsm[buf ^ 1][srow + 64][scol] = rc1;
      } else {
#pragma unroll
        for (int e = 0; e < 8; ++e) {
          Bsm[buf ^ 1][bnn + e][bkk] = rb0[e];
          Bsm[buf ^ 1][bnn + e][bkk + 16] = rb1[e];
        }
      }
#endif
    }
    __syncthreads();
  }

  // ---- epilogue: C/D layout -> lanes 0-15 M=r, lanes 16-31 M=8+r, N=lr ----
#pragma unroll
  for (int m = 0; m < 2; ++m) {
#pragma unroll
    for (int n = 0; n < 4; ++n) {
      const int gcol = nBlock + waveN * 64 + n * 16 + lr;
      const float bv = bias ? bias[gcol] : 0.0f;
#pragma unroll
      for (int r = 0; r < 8; ++r) {
        const int grow = mBlock + waveM * 32 + m * 16 + lh * 8 + r;
        const float v = acc[m][n][r] * scale + bv;
        Ob[(long long)grow * N + gcol] = (OutT)v;
      }
    }
  }
}

// ---------------------------------------------------------------------------
// Row softmax over S=4096 columns, one block per (b, q) row, in-place bf16.
// Applies attention mask (mask==0 -> -inf) before the softmax.
// ---------------------------------------------------------------------------
__global__ __launch_bounds__(BDIM)
void softmax_rows(__bf16* __restrict__ P, const int* __restrict__ mask, int S) {
  const long long row = blockIdx.x;  // 0 .. B*S-1
  __bf16* p = P + row * (long long)S;
  const int* mrow = mask + row * (long long)S;
  const int tid = threadIdx.x;

  float local[16];
  float mx = -INFINITY;
#pragma unroll
  for (int i = 0; i < 16; ++i) {
    const int idx = tid + i * BDIM;
    float v = (float)p[idx];
    if (mrow[idx] == 0) v = -INFINITY;
    local[i] = v;
    mx = fmaxf(mx, v);
  }

  __shared__ float red[BDIM];
  red[tid] = mx;
  __syncthreads();
  for (int s = BDIM / 2; s > 0; s >>= 1) {
    if (tid < s) red[tid] = fmaxf(red[tid], red[tid + s]);
    __syncthreads();
  }
  mx = red[0];
  __syncthreads();

  float sum = 0.0f;
#pragma unroll
  for (int i = 0; i < 16; ++i) {
    const float e = __expf(local[i] - mx);
    local[i] = e;
    sum += e;
  }
  red[tid] = sum;
  __syncthreads();
  for (int s = BDIM / 2; s > 0; s >>= 1) {
    if (tid < s) red[tid] += red[tid + s];
    __syncthreads();
  }
  const float inv = 1.0f / red[0];
#pragma unroll
  for (int i = 0; i < 16; ++i) p[tid + i * BDIM] = (__bf16)(local[i] * inv);
}

// ---------------------------------------------------------------------------
extern "C" void kernel_launch(void* const* d_in, const int* in_sizes, int n_in,
                              void* d_out, int out_size, void* d_ws,
                              size_t ws_size, hipStream_t stream) {
  (void)in_sizes; (void)n_in; (void)out_size; (void)ws_size;

  const float* X    = (const float*)d_in[0];
  const int*   mask = (const int*)d_in[1];
  const float* Wq   = (const float*)d_in[2];
  const float* bq   = (const float*)d_in[3];
  const float* Wk   = (const float*)d_in[4];
  const float* bk   = (const float*)d_in[5];
  const float* Wv   = (const float*)d_in[6];
  const float* bv   = (const float*)d_in[7];
  float* out = (float*)d_out;

  const long long nX = (long long)kB * kS * kE;   // 8.4M
  const long long nW = (long long)kH * kE;        // 1M
  const long long nQ = (long long)kB * kS * kH;   // 8.4M
  const long long nP = (long long)kB * kS * kS;   // 33.6M

  char* w = (char*)d_ws;
  __bf16* Xb  = (__bf16*)w;              w += nX * 2;
  __bf16* Wqb = (__bf16*)w;              w += nW * 2;
  __bf16* Wkb = (__bf16*)w;              w += nW * 2;
  __bf16* Wvb = (__bf16*)w;              w += nW * 2;
  __bf16* Qb  = (__bf16*)w;              w += nQ * 2;
  __bf16* Kb  = (__bf16*)w;              w += nQ * 2;
  __bf16* Vb  = (__bf16*)w;              w += nQ * 2;
  __bf16* Pb  = (__bf16*)w;              w += nP * 2;

  // 1) casts
  cast_f32_to_bf16<<<4096, BDIM, 0, stream>>>(X, Xb, nX);
  cast_f32_to_bf16<<<512,  BDIM, 0, stream>>>(Wq, Wqb, nW);
  cast_f32_to_bf16<<<512,  BDIM, 0, stream>>>(Wk, Wkb, nW);
  cast_f32_to_bf16<<<512,  BDIM, 0, stream>>>(Wv, Wvb, nW);

  // 2) projections: [8192,1024] x [1024,1024]^T + bias -> bf16
  {
    dim3 grid(kH / 128, (kB * kS) / 128, 1);
    wmma_gemm_bf16<true, __bf16><<<grid, BDIM, 0, stream>>>(
        Xb, Wqb, Qb, bq, 1.0f, kB * kS, kH, kE, 0, 0, 0);
    wmma_gemm_bf16<true, __bf16><<<grid, BDIM, 0, stream>>>(
        Xb, Wkb, Kb, bk, 1.0f, kB * kS, kH, kE, 0, 0, 0);
    wmma_gemm_bf16<true, __bf16><<<grid, BDIM, 0, stream>>>(
        Xb, Wvb, Vb, bv, 1.0f, kB * kS, kH, kE, 0, 0, 0);
  }

  // 3) scores: per batch [4096,1024] x [4096,1024]^T * 1/sqrt(H) -> bf16
  {
    dim3 grid(kS / 128, kS / 128, kB);
    const float scale = 0.03125f;  // 1/sqrt(1024)
    wmma_gemm_bf16<true, __bf16><<<grid, BDIM, 0, stream>>>(
        Qb, Kb, Pb, nullptr, scale, kS, kS, kH,
        (long long)kS * kH, (long long)kS * kH, (long long)kS * kS);
  }

  // 4) masked softmax rows, in place
  softmax_rows<<<kB * kS, BDIM, 0, stream>>>(Pb, mask, kS);

  // 5) out = P @ V : per batch [4096,4096] x [4096,1024] -> f32
  {
    dim3 grid(kH / 128, kS / 128, kB);
    wmma_gemm_bf16<false, float><<<grid, BDIM, 0, stream>>>(
        Pb, Vb, out, nullptr, 1.0f, kS, kH, kS,
        (long long)kS * kS, (long long)kS * kH, (long long)kS * kH);
  }
}
